// PatchSampleF_42279658062096
// MI455X (gfx1250) — matrix operand order repaired and checked by reading, not compiled, not run on hardware
//
#include <hip/hip_runtime.h>

typedef __attribute__((ext_vector_type(16))) _Float16 v16h;
typedef __attribute__((ext_vector_type(8)))  float    v8f;

#define B_    16
#define C_    512
#define HW_   4096
#define P_    2048
#define NC_   256
#define ROWS  128           // rows per workgroup
#define NT    256           // threads per workgroup (8 waves)
#define XSTR  520           // x_lds row stride in halves (512 + 8 pad, 1040B = 16B aligned)
#define HSTR  264           // h_lds row stride in halves (256 + 8 pad, 528B = 16B aligned)
#define WSTR  40            // w_lds row stride in halves (32 + 8 pad, 80B; 20-dword stride -> conflict-free)

#define LDS_X_OFF 0
#define LDS_W_OFF (ROWS * XSTR * 2)                 // 133120
#define LDS_P_OFF (LDS_W_OFF + NC_ * WSTR * 2)      // +20480
#define LDS_BYTES (LDS_P_OFF + ROWS * 4)            // +512 = 154112

// ws layout: f16 copies of the weights (converted once per launch)
#define WS_W1H_OFF 0
#define WS_W2H_OFF (NC_ * C_ * 2)                   // 262144
#define WS_BYTES   (WS_W2H_OFF + NC_ * NC_ * 2)     // 393216

// ---------------- prep: fp32 -> f16 weight conversion ----------------
__global__ __launch_bounds__(256) void cvt_f32_to_f16(
    const float* __restrict__ src, _Float16* __restrict__ dst, int n4)
{
    const int i = blockIdx.x * blockDim.x + threadIdx.x;
    if (i < n4) {
        float4 f = ((const float4*)src)[i];
        union { _Float16 h[4]; uint2 u; } u;
        u.h[0] = (_Float16)f.x; u.h[1] = (_Float16)f.y;
        u.h[2] = (_Float16)f.z; u.h[3] = (_Float16)f.w;
        ((uint2*)dst)[i] = u.u;
    }
}

// ---------------- fused gather + MLP + L2-normalize ----------------
__global__ __launch_bounds__(NT) void psf_fused(
    const float* __restrict__ feat, const int* __restrict__ patch_id,
    const _Float16* __restrict__ w1h, const float* __restrict__ b1,
    const _Float16* __restrict__ w2h, const float* __restrict__ b2,
    float* __restrict__ out)
{
    extern __shared__ char smem[];
    _Float16* xh  = (_Float16*)(smem + LDS_X_OFF);   // x tile, later aliased as h tile
    _Float16* wl  = (_Float16*)(smem + LDS_W_OFF);   // staged weight K-slice (f16)
    int*      pid = (int*)     (smem + LDS_P_OFF);

    const int tid  = threadIdx.x;
    const int lane = tid & 31;
    const int wave = tid >> 5;
    const int l15  = lane & 15;
    const int lh   = lane >> 4;              // 0: lanes 0-15, 1: lanes 16-31
    const int r0   = blockIdx.x * ROWS;      // global output row base
    const int b    = r0 / P_;                // batch index (128 | 2048 -> constant per block)
    const int p0   = r0 % P_;

    // LDS byte offset of the weight staging buffer (for async global->LDS)
    const unsigned wl_off = (unsigned)(size_t)(void*)wl;

    // ---- stage patch ids for this block's 128 rows ----
    if (tid < ROWS) pid[tid] = patch_id[p0 + tid];
    __syncthreads();

    // ---- gather x[128][512] (f32 strided gather) -> f16 LDS ----
    // thread t: fixed row r = t&127 (fixed hw), channels c = 2*i + (t>>7).
    // Each vector load's 32 lanes read random offsets inside ONE 16KB (b,c) plane.
    {
        const float* fb = feat + (size_t)b * C_ * HW_;
        const int r  = tid & 127;
        const int c0 = tid >> 7;
        const int hw = pid[r];
        #pragma unroll 8
        for (int i = 0; i < 256; ++i) {
            const int c = i * 2 + c0;
            xh[r * XSTR + c] = (_Float16)fb[(size_t)c * HW_ + hw];
        }
    }

    // ---- per-lane bias copies: column n = j*16 + l15 ----
    float bias1[16], bias2[16];
    #pragma unroll
    for (int j = 0; j < 16; ++j) {
        bias1[j] = b1[j * 16 + l15];
        bias2[j] = b2[j * 16 + l15];
    }
    __syncthreads();

    const int myrow = wave * 16 + l15;       // A-operand row this lane serves
    v8f zero = {};
    v8f acc[16];
    #pragma unroll
    for (int j = 0; j < 16; ++j) acc[j] = zero;

    // ================= GEMM1: h = relu(x @ w1^T + b1), K = 512 =================
    for (int k = 0; k < 16; ++k) {
        // async-stage w1h[n = tid][k*32 .. k*32+31] (64B per row) into LDS, 80B row stride
        {
            const unsigned long long ga =
                (unsigned long long)(const void*)w1h + (size_t)tid * (C_ * 2) + (size_t)k * 64;
            const unsigned dstb = wl_off + tid * (WSTR * 2);
            #pragma unroll
            for (int ch = 0; ch < 4; ++ch) {
                unsigned            d = dstb + ch * 16;
                unsigned long long  g = ga   + ch * 16;
                asm volatile("global_load_async_to_lds_b128 %0, %1, off"
                             :: "v"(d), "v"(g) : "memory");
            }
            asm volatile("s_wait_asynccnt 0" ::: "memory");
        }
        __syncthreads();

        // A operand: ISA 16-bit A layout: lane<16 -> K {0..7,16..23}, lane>=16 -> K {8..15,24..31}
        union { v16h v; uint4 q[2]; } A;
        {
            const _Float16* ap = xh + myrow * XSTR + k * 32 + lh * 8;
            A.q[0] = *(const uint4*)(ap);
            A.q[1] = *(const uint4*)(ap + 16);
        }
        #pragma unroll
        for (int j = 0; j < 16; ++j) {
            // B operand: B[k][n] = w1[n][k]; lane<16 -> K {0..7,16..23}, lane>=16 -> K {8..15,24..31} (matches A)
            union { v16h v; uint4 q[2]; } Bv;
            const _Float16* bp = wl + (j * 16 + l15) * WSTR + lh * 8;
            Bv.q[0] = *(const uint4*)(bp);
            Bv.q[1] = *(const uint4*)(bp + 16);
            acc[j] = __builtin_amdgcn_wmma_f32_16x16x32_f16(
                false, A.v, false, Bv.v, (short)0, acc[j], false, false);
        }
        __syncthreads();
    }

    // ---- bias + ReLU, write h (f16) into LDS (aliases x buffer; barrier above protects) ----
    #pragma unroll
    for (int j = 0; j < 16; ++j) {
        #pragma unroll
        for (int m = 0; m < 8; ++m) {
            float v = acc[j][m] + bias1[j];
            v = v > 0.0f ? v : 0.0f;
            const int row = wave * 16 + m + lh * 8;      // D layout: lane<16 rows 0-7, lane>=16 rows 8-15
            xh[row * HSTR + j * 16 + l15] = (_Float16)v;
        }
    }

    #pragma unroll
    for (int j = 0; j < 16; ++j) acc[j] = zero;

    // ================= GEMM2: y = h @ w2^T + b2, K = 256 =================
    for (int k = 0; k < 8; ++k) {
        {
            const unsigned long long ga =
                (unsigned long long)(const void*)w2h + (size_t)tid * (NC_ * 2) + (size_t)k * 64;
            const unsigned dstb = wl_off + tid * (WSTR * 2);
            #pragma unroll
            for (int ch = 0; ch < 4; ++ch) {
                unsigned            d = dstb + ch * 16;
                unsigned long long  g = ga   + ch * 16;
                asm volatile("global_load_async_to_lds_b128 %0, %1, off"
                             :: "v"(d), "v"(g) : "memory");
            }
            asm volatile("s_wait_asynccnt 0" ::: "memory");
        }
        __syncthreads();   // also orders the h stores above vs. h reads below (first iter)

        union { v16h v; uint4 q[2]; } A;
        {
            const _Float16* ap = xh + myrow * HSTR + k * 32 + lh * 8;
            A.q[0] = *(const uint4*)(ap);
            A.q[1] = *(const uint4*)(ap + 16);
        }
        #pragma unroll
        for (int j = 0; j < 16; ++j) {
            union { v16h v; uint4 q[2]; } Bv;
            const _Float16* bp = wl + (j * 16 + l15) * WSTR + lh * 8;
            Bv.q[0] = *(const uint4*)(bp);
            Bv.q[1] = *(const uint4*)(bp + 16);
            acc[j] = __builtin_amdgcn_wmma_f32_16x16x32_f16(
                false, A.v, false, Bv.v, (short)0, acc[j], false, false);
        }
        __syncthreads();
    }

    // ---- bias2, row-wise L2 norm, scale, store fp32 ----
    float ss[8];
    #pragma unroll
    for (int m = 0; m < 8; ++m) ss[m] = 0.0f;
    #pragma unroll
    for (int j = 0; j < 16; ++j) {
        #pragma unroll
        for (int m = 0; m < 8; ++m) {
            float v = acc[j][m] + bias2[j];
            acc[j][m] = v;
            ss[m] += v * v;
        }
    }
    // row m's 256 values live across lanes 0-15 (rows 0-7) / lanes 16-31 (rows 8-15):
    // xor-reduce within each 16-lane half.
    #pragma unroll
    for (int m = 0; m < 8; ++m) {
        ss[m] += __shfl_xor(ss[m], 1, 32);
        ss[m] += __shfl_xor(ss[m], 2, 32);
        ss[m] += __shfl_xor(ss[m], 4, 32);
        ss[m] += __shfl_xor(ss[m], 8, 32);
        ss[m] = 1.0f / (sqrtf(ss[m]) + 1e-7f);
    }
    #pragma unroll
    for (int j = 0; j < 16; ++j) {
        #pragma unroll
        for (int m = 0; m < 8; ++m) {
            const int row = r0 + wave * 16 + m + lh * 8;
            out[(size_t)row * NC_ + j * 16 + l15] = acc[j][m] * ss[m];
        }
    }
}

extern "C" void kernel_launch(void* const* d_in, const int* in_sizes, int n_in,
                              void* d_out, int out_size, void* d_ws, size_t ws_size,
                              hipStream_t stream) {
    const float* feat     = (const float*)d_in[0];
    const int*   patch_id = (const int*)  d_in[1];
    const float* w1       = (const float*)d_in[2];
    const float* b1       = (const float*)d_in[3];
    const float* w2       = (const float*)d_in[4];
    const float* b2       = (const float*)d_in[5];
    float*       out      = (float*)d_out;
    (void)in_sizes; (void)n_in; (void)out_size; (void)ws_size;

    _Float16* w1h = (_Float16*)((char*)d_ws + WS_W1H_OFF);
    _Float16* w2h = (_Float16*)((char*)d_ws + WS_W2H_OFF);

    // one-time (per launch) fp32 -> f16 weight conversion into workspace
    {
        const int n4_1 = (NC_ * C_) / 4;   // 32768
        const int n4_2 = (NC_ * NC_) / 4;  // 16384
        cvt_f32_to_f16<<<(n4_1 + 255) / 256, 256, 0, stream>>>(w1, w1h, n4_1);
        cvt_f32_to_f16<<<(n4_2 + 255) / 256, 256, 0, stream>>>(w2, w2h, n4_2);
    }

    hipFuncSetAttribute((const void*)psf_fused,
                        hipFuncAttributeMaxDynamicSharedMemorySize, LDS_BYTES);

    const int nblocks = (B_ * P_) / ROWS;    // 32768 / 128 = 256
    psf_fused<<<nblocks, NT, LDS_BYTES, stream>>>(feat, patch_id, w1h, b1, w2h, b2, out);
}
